// LSTMModule_34729105555540
// MI455X (gfx1250) — matrix operand order repaired and checked by reading, not compiled, not run on hardware
//
#include <hip/hip_runtime.h>

// ---------------------------------------------------------------------------
// LSTM (T=1024, B=128, I=512, H=512) for MI455X / gfx1250, wave32 + WMMA.
//
//  * batch rows are recurrence-independent -> 8 workgroups of 16 batch rows
//    run the sequential scan with NO cross-workgroup sync.
//  * gate GEMMs use v_wmma_f32_16x16x32_bf16 (f32 accumulate, 8x the K/instr
//    of f32 WMMA); W/R converted once to bf16 (4 MB, L2-resident all run).
//  * wave w owns hidden tile w and computes its own i/o/f/ct gate tiles, so
//    the nonlinear cell update is fully in registers; the cell state c is a
//    persistent 8-VGPR accumulator across all 1024 steps.
//  * If d_ws is large enough (512 MB), the recurrence-independent X.W^T half
//    is hoisted into a whole-GPU pre-GEMM (8192 blocks) that stores
//    bias + X.W^T in WMMA C-tile order (bf16), halving the K of the
//    sequential scan (the critical path) to 512.
// ---------------------------------------------------------------------------

typedef __attribute__((ext_vector_type(16))) __bf16 bf16x16;
typedef __attribute__((ext_vector_type(8)))  __bf16 bf16x8;
typedef __attribute__((ext_vector_type(8)))  float  f32x8;
typedef __attribute__((ext_vector_type(4)))  float  f32x4;

#define T_STEPS 1024
#define BATCH   128
#define INSZ    512
#define HID     512
#define GATES   2048          // 4*HID
#define XS_STRIDE 520         // padded LDS row stride (bf16 elems): bank-conflict free

__device__ __forceinline__ float fast_sigmoid(float x) {
  return __builtin_amdgcn_rcpf(1.0f + __expf(-x));
}

__device__ __forceinline__ float fast_tanh(float x) {
#if __has_builtin(__builtin_amdgcn_tanhf)
  return __builtin_amdgcn_tanhf(x);
#else
  float e = __expf(-2.0f * x);
  return (1.0f - e) * __builtin_amdgcn_rcpf(1.0f + e);
#endif
}

// Build a 16x32 bf16 WMMA A/B fragment from two 8-element chunks.
__device__ __forceinline__ bf16x16 make_frag(const __bf16* p0, const __bf16* p1) {
  bf16x8 lo = *(const bf16x8*)p0;
  bf16x8 hi = *(const bf16x8*)p1;
  return __builtin_shufflevector(lo, hi, 0,1,2,3,4,5,6,7,8,9,10,11,12,13,14,15);
}

// Stage one 16x512 f32 row-block of X into LDS as bf16 (A-operand layout).
__device__ __forceinline__ void stage_x_tile(const float* __restrict__ src,
                                             __bf16* __restrict__ xs,
                                             int sm, int sk) {
  const float* p = src + (size_t)sm * INSZ + sk;
  f32x4 v0 = *(const f32x4*)(p);
  f32x4 v1 = *(const f32x4*)(p + 4);
  bf16x8 pk;
  pk[0] = (__bf16)v0[0]; pk[1] = (__bf16)v0[1];
  pk[2] = (__bf16)v0[2]; pk[3] = (__bf16)v0[3];
  pk[4] = (__bf16)v1[0]; pk[5] = (__bf16)v1[1];
  pk[6] = (__bf16)v1[2]; pk[7] = (__bf16)v1[3];
  *(bf16x8*)(xs + sm * XS_STRIDE + sk) = pk;
}

// One-time: f32 -> bf16 for W and R; fold the two bias halves together.
__global__ void lstm_convert_kernel(const float* __restrict__ W,
                                    const float* __restrict__ R,
                                    const float* __restrict__ Bf,
                                    __bf16* __restrict__ Wb,
                                    __bf16* __restrict__ Rb,
                                    float* __restrict__ bias) {
  int i = blockIdx.x * blockDim.x + threadIdx.x;
  if (i < GATES * INSZ) {
    Wb[i] = (__bf16)W[i];
    Rb[i] = (__bf16)R[i];
  }
  if (i < GATES) bias[i] = Bf[i] + Bf[i + GATES];
}

// ---------------------------------------------------------------------------
// Phase 1 (whole GPU): xg[t,wg,wave,g] = bias + X[t] . W^T  in bf16,
// stored in the exact WMMA C-tile register order used by the scan kernel.
// Grid: T_STEPS*8 blocks of 1024 threads.
// ---------------------------------------------------------------------------
__global__ __launch_bounds__(1024, 1) void lstm_xgemm_kernel(
    const float*  __restrict__ X,
    const __bf16* __restrict__ Wb,
    const float*  __restrict__ bias,
    __bf16*       __restrict__ xg) {
  __shared__ __bf16 xs[16 * XS_STRIDE];

  const int tid  = threadIdx.x;
  const int wave = tid >> 5;
  const int lane = tid & 31;
  const int ln   = lane & 15;
  const int hi   = (lane >> 4) & 1;
  const int t    = blockIdx.x >> 3;
  const int wg   = blockIdx.x & 7;

  const int sm = tid >> 6;
  const int sk = (tid & 63) * 8;
  stage_x_tile(X + ((size_t)t * BATCH + wg * 16) * INSZ, xs, sm, sk);

  const int nh = wave * 16 + ln;
  const int koffA = 16 * hi;
  const int koffB = 16 * hi;
  const __bf16* wp[4];
  float bb[4];
  #pragma unroll
  for (int g = 0; g < 4; ++g) {
    wp[g] = Wb + (size_t)(g * HID + nh) * INSZ + koffB;
    bb[g] = bias[g * HID + nh];
  }
  const __bf16* xrow = xs + ln * XS_STRIDE + koffA;

  f32x8 acc[4];
  #pragma unroll
  for (int g = 0; g < 4; ++g)
    acc[g] = (f32x8){bb[g],bb[g],bb[g],bb[g],bb[g],bb[g],bb[g],bb[g]};

  __syncthreads();

  #pragma unroll 2
  for (int kf = 0; kf < 16; ++kf) {
    const int k0 = kf * 32;
    bf16x16 A = make_frag(xrow + k0, xrow + k0 + 8);
    #pragma unroll
    for (int g = 0; g < 4; ++g) {
      bf16x16 B = make_frag(wp[g] + k0, wp[g] + k0 + 8);
      acc[g] = __builtin_amdgcn_wmma_f32_16x16x32_bf16(false, A, false, B,
                                                       (short)0, acc[g], false, false);
    }
  }

  // store in C-tile register order: [(t*8+wg)*32+wave][g][lane][8] bf16
  __bf16* dst = xg + ((((size_t)t * 8 + wg) * 32 + wave) * 4) * 256 + (size_t)lane * 8;
  #pragma unroll
  for (int g = 0; g < 4; ++g) {
    bf16x8 o;
    #pragma unroll
    for (int e = 0; e < 8; ++e) o[e] = (__bf16)acc[g][e];
    *(bf16x8*)(dst + (size_t)g * 256) = o;
  }
}

// ---------------------------------------------------------------------------
// Sequential scan. PRE=true: gate seed loaded from precomputed xg (K=512).
// PRE=false: fused X.W^T inside the loop (K=1024), no big workspace needed.
// Grid: 8 blocks of 1024 threads (one WGP per 16 batch rows).
// ---------------------------------------------------------------------------
template <bool PRE>
__global__ __launch_bounds__(1024, 1) void lstm_scan_kernel(
    const float*  __restrict__ X,
    const __bf16* __restrict__ xg,
    const __bf16* __restrict__ Wb,
    const __bf16* __restrict__ Rb,
    const float*  __restrict__ bias,
    float*        __restrict__ out) {
  __shared__ __bf16 hs[16 * XS_STRIDE];           // h, bf16
  __shared__ __bf16 xs[PRE ? 16 : 16 * XS_STRIDE]; // X[t] staging (fused only)

  const int tid  = threadIdx.x;          // 0..1023 (32 waves of 32)
  const int wave = tid >> 5;             // hidden tile id 0..31
  const int lane = tid & 31;
  const int ln   = lane & 15;
  const int hi   = (lane >> 4) & 1;
  const int wg   = blockIdx.x;           // batch group 0..7

  const int sm = tid >> 6;
  const int sk = (tid & 63) * 8;
  { // h := 0
    bf16x8 z = {};
    *(bf16x8*)(hs + sm * XS_STRIDE + sk) = z;
  }

  const int nh = wave * 16 + ln;         // hidden col 0..511
  const int koffA = 16 * hi;
  const int koffB = 16 * hi;

  const __bf16* wp[4];
  const __bf16* rp[4];
  float bb[4];
  #pragma unroll
  for (int g = 0; g < 4; ++g) {
    wp[g] = Wb + (size_t)(g * HID + nh) * INSZ + koffB;
    rp[g] = Rb + (size_t)(g * HID + nh) * HID + koffB;
    bb[g] = bias[g * HID + nh];
  }
  const __bf16* xrow = xs + ln * XS_STRIDE + koffA;
  const __bf16* hrow = hs + ln * XS_STRIDE + koffA;
  // this wave's slice of the precomputed gate seed (C-tile order)
  const __bf16* xgw = PRE ? xg + (((size_t)wg * 32 + wave) * 4) * 256 + (size_t)lane * 8
                          : nullptr;
  const size_t xg_tstride = (size_t)8 * 32 * 4 * 256;   // elements per time step

  f32x8 creg = {};                       // persistent cell state (16x16 tile)
  __syncthreads();

  #pragma unroll 1
  for (int t = 0; t < T_STEPS; ++t) {
    f32x8 acc[4];

    if constexpr (PRE) {
      // gate seed = bias + X[t].W^T, precomputed, perfectly coalesced loads
      const __bf16* s = xgw + (size_t)t * xg_tstride;
      #pragma unroll
      for (int g = 0; g < 4; ++g) {
        bf16x8 v = *(const bf16x8*)(s + (size_t)g * 256);
        #pragma unroll
        for (int e = 0; e < 8; ++e) acc[g][e] = (float)v[e];
      }
      if (t + 1 < T_STEPS) __builtin_prefetch(s + xg_tstride, 0, 1);
    } else {
      // stage X[t] (16x512 f32 -> bf16 in LDS), prefetch X[t+1]
      const float* src = X + ((size_t)t * BATCH + wg * 16) * INSZ;
      stage_x_tile(src, xs, sm, sk);
      if (t + 1 < T_STEPS)
        __builtin_prefetch(src + (size_t)BATCH * INSZ + sm * INSZ + sk, 0, 1);
      __syncthreads();

      #pragma unroll
      for (int g = 0; g < 4; ++g)
        acc[g] = (f32x8){bb[g],bb[g],bb[g],bb[g],bb[g],bb[g],bb[g],bb[g]};

      #pragma unroll 2
      for (int kf = 0; kf < 16; ++kf) {        // gates += X[t] @ W^T
        const int k0 = kf * 32;
        bf16x16 A = make_frag(xrow + k0, xrow + k0 + 8);
        #pragma unroll
        for (int g = 0; g < 4; ++g) {
          bf16x16 B = make_frag(wp[g] + k0, wp[g] + k0 + 8);
          acc[g] = __builtin_amdgcn_wmma_f32_16x16x32_bf16(false, A, false, B,
                                                           (short)0, acc[g], false, false);
        }
      }
    }

    #pragma unroll 2
    for (int kf = 0; kf < 16; ++kf) {          // gates += h @ R^T
      const int k0 = kf * 32;
      bf16x16 A = make_frag(hrow + k0, hrow + k0 + 8);
      #pragma unroll
      for (int g = 0; g < 4; ++g) {
        bf16x16 B = make_frag(rp[g] + k0, rp[g] + k0 + 8);
        acc[g] = __builtin_amdgcn_wmma_f32_16x16x32_bf16(false, A, false, B,
                                                         (short)0, acc[g], false, false);
      }
    }
    __syncthreads();   // everyone done reading hs before it is overwritten

    // ---- nonlinear cell update, fully in registers ----
    // C/D tile layout: VGPR e, lane: row m = e + 8*hi, col = nh
    #pragma unroll
    for (int e = 0; e < 8; ++e) {
      float si = fast_sigmoid(acc[0][e]);
      float so = fast_sigmoid(acc[1][e]);
      float sf = fast_sigmoid(acc[2][e]);
      float ct = fast_tanh(acc[3][e]);
      float c  = sf * creg[e] + si * ct;
      creg[e]  = c;
      float h  = so * fast_tanh(c);
      const int m = e + 8 * hi;
      hs[m * XS_STRIDE + nh] = (__bf16)h;
      if (t == T_STEPS - 1) {
        out[(size_t)(wg * 16 + m) * HID + nh] = h;                       // final h
        out[(size_t)BATCH * HID + (size_t)(wg * 16 + m) * HID + nh] = c; // final c
      }
    }
    __syncthreads();   // new h visible before next step's A-fragment reads
  }
}

extern "C" void kernel_launch(void* const* d_in, const int* in_sizes, int n_in,
                              void* d_out, int out_size, void* d_ws, size_t ws_size,
                              hipStream_t stream) {
  const float* X  = (const float*)d_in[0];  // (1024,128,512)
  const float* W  = (const float*)d_in[1];  // (1,2048,512)
  const float* R  = (const float*)d_in[2];  // (1,2048,512)
  const float* Bf = (const float*)d_in[3];  // (1,4096)

  char* ws = (char*)d_ws;
  __bf16* Wb   = (__bf16*)(ws);                             // 2 MB
  __bf16* Rb   = (__bf16*)(ws + (size_t)2 * 1024 * 1024);   // 2 MB
  float*  bsum = (float*)(ws + (size_t)4 * 1024 * 1024);    // 8 KB
  const size_t base = (size_t)4 * 1024 * 1024 + 8 * 1024;
  const size_t xg_bytes = (size_t)T_STEPS * BATCH * GATES * sizeof(__bf16); // 512 MB
  __bf16* xg = (__bf16*)(ws + base);

  lstm_convert_kernel<<<(GATES * INSZ + 255) / 256, 256, 0, stream>>>(W, R, Bf, Wb, Rb, bsum);

  if (ws_size >= base + xg_bytes) {
    // whole-GPU pre-GEMM of the recurrence-independent half, then light scan
    lstm_xgemm_kernel<<<T_STEPS * 8, 1024, 0, stream>>>(X, Wb, bsum, xg);
    lstm_scan_kernel<true><<<BATCH / 16, 1024, 0, stream>>>(X, xg, Wb, Rb, bsum, (float*)d_out);
  } else {
    // fused fallback: no large workspace required (~4 MB)
    lstm_scan_kernel<false><<<BATCH / 16, 1024, 0, stream>>>(X, nullptr, Wb, Rb, bsum, (float*)d_out);
  }
}
